// NeighborComm_39582418600050
// MI455X (gfx1250) — matrix operand order repaired and checked by reading, not compiled, not run on hardware
//
#include <hip/hip_runtime.h>
#include <hip/hip_bf16.h>
#include <math.h>

// Problem constants from the reference (B=8, N=4096, D=64, P=3, K=6).
#define BB 8
#define NN 4096
#define DD 64
#define WSTRIDE 35   // LDS row stride (u32 units); avoids all bank conflicts for
                     // both A-frag (+4*half) and B-frag (+4*half) read patterns.

#define AS1 __attribute__((address_space(1)))
#define AS3 __attribute__((address_space(3)))

// Async global->LDS staging (ASYNCcnt path), with sync fallback.
#if defined(__has_builtin)
#if __has_builtin(__builtin_amdgcn_global_load_async_to_lds_b32) && \
    __has_builtin(__builtin_amdgcn_s_wait_asynccnt)
#define USE_ASYNC_STAGE 1
#endif
#endif
#ifndef USE_ASYNC_STAGE
#define USE_ASYNC_STAGE 0
#endif

typedef __attribute__((ext_vector_type(16))) __bf16 v16bf;
typedef __attribute__((ext_vector_type(8)))  float  v8f;

union Frag { unsigned u[8]; v16bf v; };

static __device__ __forceinline__ unsigned pack2bf(float lo, float hi) {
  __hip_bfloat16 a = __float2bfloat16(lo);
  __hip_bfloat16 b = __float2bfloat16(hi);
  unsigned short ua = *reinterpret_cast<unsigned short*>(&a);
  unsigned short ub = *reinterpret_cast<unsigned short*>(&b);
  return ((unsigned)ub << 16) | (unsigned)ua;
}

// ---------------------------------------------------------------------------
// Kernel 1: Q/K/V projections via v_wmma_f32_16x16x32_bf16.
// Block = 128 threads (4 waves). Each block: batch bi, 64 rows of h.
// Each wave owns a 16-column tile; Q,K,V share the A fragment.
// ---------------------------------------------------------------------------
__global__ void __launch_bounds__(128) qkv_proj_wmma(
    const float* __restrict__ h,
    const float* __restrict__ Wq, const float* __restrict__ bq,
    const float* __restrict__ Wk, const float* __restrict__ bk,
    const float* __restrict__ Wv, const float* __restrict__ bv,
    float* __restrict__ Q, float* __restrict__ K, float* __restrict__ V)
{
  __shared__ unsigned w_s[3][DD][WSTRIDE];  // W[d][k] as packed bf16 pairs
  __shared__ unsigned h_s[16][WSTRIDE];     // h tile as packed bf16 pairs
  __shared__ float    b_s[3][DD];

  const int bi  = blockIdx.y;
  const int tid = threadIdx.x;
  const int rowChunk = blockIdx.x * 64;

  // Stage Wq/Wk/Wv (64x64 f32 each) as bf16 pairs.
  for (int t = tid; t < 3 * 64 * 32; t += 128) {
    int m = t >> 11, rem = t & 2047;
    int row = rem >> 5, kp = rem & 31;
    const float2* src = (m == 0) ? (const float2*)Wq
                      : (m == 1) ? (const float2*)Wk : (const float2*)Wv;
    float2 w = src[(row << 5) + kp];
    w_s[m][row][kp] = pack2bf(w.x, w.y);
  }
  if (tid < 192) {
    int m = tid >> 6, c = tid & 63;
    const float* src = (m == 0) ? bq : (m == 1) ? bk : bv;
    b_s[m][c] = src[c];
  }

  const int lane = tid & 31;
  const int wid  = tid >> 5;
  const int col0 = wid * 16;
  const int half = lane >> 4;   // ISA 16-bit fragment half-wave split
  const int m16  = lane & 15;

  for (int rt = 0; rt < 4; ++rt) {
    const int row0 = rowChunk + rt * 16;
    __syncthreads();  // previous iteration done reading h_s (and W staged, rt==0)
    for (int t = tid; t < 16 * 32; t += 128) {
      int row = t >> 5, kp = t & 31;
      const float2* hp = (const float2*)(h + ((size_t)bi * NN + row0 + row) * DD);
      float2 hv = hp[kp];
      h_s[row][kp] = pack2bf(hv.x, hv.y);
    }
    __syncthreads();

    v8f cq = {}; v8f ck = {}; v8f cv = {};
#pragma unroll
    for (int kk = 0; kk < 2; ++kk) {   // K = 64 = 2 x 32
      const int kb = kk * 16;          // u32-pair units
      Frag a, fq, fk, fv;
#pragma unroll
      for (int v = 0; v < 8; ++v) {
        // A 16x32 bf16 layout: v0..3 -> K 0..7 (+8*half), v4..7 -> K 16..23 (+8*half)
        int ai = kb + (v & 3) + ((v >> 2) << 3) + 4 * half;
        a.u[v] = h_s[m16][ai];
        // B 32x16 bf16 layout: same K order as A (v0..3 -> K 0..7 (+8*half), v4..7 -> K 16..23 (+8*half)); B[k][n] = W[n][k]
        int bi2 = kb + (v & 3) + ((v >> 2) << 3) + 4 * half;
        fq.u[v] = w_s[0][col0 + m16][bi2];
        fk.u[v] = w_s[1][col0 + m16][bi2];
        fv.u[v] = w_s[2][col0 + m16][bi2];
      }
      cq = __builtin_amdgcn_wmma_f32_16x16x32_bf16(false, a.v, false, fq.v, (short)0, cq, false, false);
      ck = __builtin_amdgcn_wmma_f32_16x16x32_bf16(false, a.v, false, fk.v, (short)0, ck, false, false);
      cv = __builtin_amdgcn_wmma_f32_16x16x32_bf16(false, a.v, false, fv.v, (short)0, cv, false, false);
    }

    const int col = col0 + m16;
    const float bqv = b_s[0][col], bkv = b_s[1][col], bvv = b_s[2][col];
    const size_t obase = ((size_t)bi * NN + row0) * DD;
#pragma unroll
    for (int r = 0; r < 8; ++r) {   // D layout: VGPR r -> M = r + 8*half, N = m16
      const size_t o = obase + (size_t)(r + 8 * half) * DD + col;
      Q[o] = cq[r] + bqv;
      K[o] = ck[r] + bkv;
      V[o] = cv[r] + bvv;
    }
  }
}

// ---------------------------------------------------------------------------
// Kernel 2: fused O(N^2) KNN (top-6 by squared distance, stable tie-break)
// + 6-neighbor attention. One lane per node. pos[b] staged in LDS as padded
// float4 so the candidate scan is a single broadcast ds_load_b128 per j.
// Staging uses GLOBAL_LOAD_ASYNC_TO_LDS (ASYNCcnt) when available.
// ---------------------------------------------------------------------------
__global__ void __launch_bounds__(256) knn_attn(
    const float* __restrict__ pos,
    const float* __restrict__ Qm, const float* __restrict__ Km,
    const float* __restrict__ Vm, float* __restrict__ out)
{
  __shared__ float4 p4s[NN];   // 64 KB of the 320 KB WGP LDS ({x,y,z,pad})
  const int bi  = blockIdx.y;
  const int tid = threadIdx.x;

  const float* pb = pos + (size_t)bi * NN * 3;
#if USE_ASYNC_STAGE
  for (int t = tid; t < NN; t += 256) {
    AS1 int* g = (AS1 int*)(pb + 3 * t);      // builtin wants global int*
    AS3 int* l = (AS3 int*)(&p4s[t]);         // and shared int*
    __builtin_amdgcn_global_load_async_to_lds_b32(g,     l,     0, 0);
    __builtin_amdgcn_global_load_async_to_lds_b32(g + 1, l + 1, 0, 0);
    __builtin_amdgcn_global_load_async_to_lds_b32(g + 2, l + 2, 0, 0);
  }
  __builtin_amdgcn_s_wait_asynccnt(0);
#else
  for (int t = tid; t < NN * 3; t += 256) {   // coalesced global read
    float f = pb[t];
    int j = t / 3, c = t - 3 * j;
    if (c == 0) p4s[j].x = f; else if (c == 1) p4s[j].y = f; else p4s[j].z = f;
  }
#endif
  __syncthreads();

  const int i = blockIdx.x * 256 + tid;
  const float4 me = p4s[i];
  const float mx = me.x, my = me.y, mz = me.z;

  float nd0 = 3.0e38f, nd1 = 3.0e38f, nd2 = 3.0e38f,
        nd3 = 3.0e38f, nd4 = 3.0e38f, nd5 = 3.0e38f;
  int j0 = 0, j1 = 0, j2 = 0, j3 = 0, j4 = 0, j5 = 0;

#pragma unroll 4
  for (int j = 0; j < NN; ++j) {
    float4 pj = p4s[j];                       // one broadcast ds_load_b128
    float dx = pj.x - mx, dy = pj.y - my, dz = pj.z - mz;
    float q2 = dx * dx; q2 = fmaf(dy, dy, q2); q2 = fmaf(dz, dz, q2);
    if (q2 < nd5) {           // rare (~1%): branchless sorted insert
      bool b0 = q2 < nd0, b1 = q2 < nd1, b2 = q2 < nd2, b3 = q2 < nd3, b4 = q2 < nd4;
      nd5 = b4 ? nd4 : q2;                j5 = b4 ? j4 : j;
      nd4 = b3 ? nd3 : (b4 ? q2 : nd4);   j4 = b3 ? j3 : (b4 ? j : j4);
      nd3 = b2 ? nd2 : (b3 ? q2 : nd3);   j3 = b2 ? j2 : (b3 ? j : j3);
      nd2 = b1 ? nd1 : (b2 ? q2 : nd2);   j2 = b1 ? j1 : (b2 ? j : j2);
      nd1 = b0 ? nd0 : (b1 ? q2 : nd1);   j1 = b0 ? j0 : (b1 ? j : j1);
      nd0 = b0 ? q2  : nd0;               j0 = b0 ? j  : j0;
    }
  }

  // ---- attention over the 6 neighbors (K/V rows are L2-resident) ----
  const size_t rowOff = ((size_t)bi * NN + i) * DD;
  const float4* qp = (const float4*)(Qm + rowOff);
  const float* kb = Km + (size_t)bi * NN * DD;
  const float* vb = Vm + (size_t)bi * NN * DD;
  const float4* k0 = (const float4*)(kb + (size_t)j0 * DD);
  const float4* k1 = (const float4*)(kb + (size_t)j1 * DD);
  const float4* k2 = (const float4*)(kb + (size_t)j2 * DD);
  const float4* k3 = (const float4*)(kb + (size_t)j3 * DD);
  const float4* k4 = (const float4*)(kb + (size_t)j4 * DD);
  const float4* k5 = (const float4*)(kb + (size_t)j5 * DD);
  const float4* v0 = (const float4*)(vb + (size_t)j0 * DD);
  const float4* v1 = (const float4*)(vb + (size_t)j1 * DD);
  const float4* v2 = (const float4*)(vb + (size_t)j2 * DD);
  const float4* v3 = (const float4*)(vb + (size_t)j3 * DD);
  const float4* v4 = (const float4*)(vb + (size_t)j4 * DD);
  const float4* v5 = (const float4*)(vb + (size_t)j5 * DD);
  __builtin_prefetch(k0, 0, 1); __builtin_prefetch(k1, 0, 1);
  __builtin_prefetch(k2, 0, 1); __builtin_prefetch(k3, 0, 1);
  __builtin_prefetch(k4, 0, 1); __builtin_prefetch(k5, 0, 1);
  __builtin_prefetch(v0, 0, 1); __builtin_prefetch(v1, 0, 1);
  __builtin_prefetch(v2, 0, 1); __builtin_prefetch(v3, 0, 1);
  __builtin_prefetch(v4, 0, 1); __builtin_prefetch(v5, 0, 1);

  float a0 = 0.f, a1 = 0.f, a2 = 0.f, a3 = 0.f, a4 = 0.f, a5 = 0.f;
#pragma unroll 4
  for (int d = 0; d < DD / 4; ++d) {
    float4 q = qp[d];
    float4 t;
    t = k0[d]; a0 += q.x * t.x + q.y * t.y + q.z * t.z + q.w * t.w;
    t = k1[d]; a1 += q.x * t.x + q.y * t.y + q.z * t.z + q.w * t.w;
    t = k2[d]; a2 += q.x * t.x + q.y * t.y + q.z * t.z + q.w * t.w;
    t = k3[d]; a3 += q.x * t.x + q.y * t.y + q.z * t.z + q.w * t.w;
    t = k4[d]; a4 += q.x * t.x + q.y * t.y + q.z * t.z + q.w * t.w;
    t = k5[d]; a5 += q.x * t.x + q.y * t.y + q.z * t.z + q.w * t.w;
  }
  const float sc = 0.125f;  // 1/sqrt(64)
  float s0 = a0 * sc, s1 = a1 * sc, s2 = a2 * sc, s3 = a3 * sc, s4 = a4 * sc, s5 = a5 * sc;
  float mM = fmaxf(fmaxf(fmaxf(s0, s1), fmaxf(s2, s3)), fmaxf(s4, s5));
  float e0 = __expf(s0 - mM), e1 = __expf(s1 - mM), e2 = __expf(s2 - mM);
  float e3 = __expf(s3 - mM), e4 = __expf(s4 - mM), e5 = __expf(s5 - mM);
  float inv = 1.0f / (e0 + e1 + e2 + e3 + e4 + e5);
  float w0 = e0 * inv, w1 = e1 * inv, w2 = e2 * inv, w3 = e3 * inv, w4 = e4 * inv, w5 = e5 * inv;

  float4* op = (float4*)(out + rowOff);
#pragma unroll 4
  for (int d = 0; d < DD / 4; ++d) {
    float4 t0 = v0[d], t1 = v1[d], t2 = v2[d], t3 = v3[d], t4 = v4[d], t5 = v5[d];
    float4 o;
    o.x = w0 * t0.x + w1 * t1.x + w2 * t2.x + w3 * t3.x + w4 * t4.x + w5 * t5.x;
    o.y = w0 * t0.y + w1 * t1.y + w2 * t2.y + w3 * t3.y + w4 * t4.y + w5 * t5.y;
    o.z = w0 * t0.z + w1 * t1.z + w2 * t2.z + w3 * t3.z + w4 * t4.z + w5 * t5.z;
    o.w = w0 * t0.w + w1 * t1.w + w2 * t2.w + w3 * t3.w + w4 * t4.w + w5 * t5.w;
    op[d] = o;
  }
}

// ---------------------------------------------------------------------------
extern "C" void kernel_launch(void* const* d_in, const int* in_sizes, int n_in,
                              void* d_out, int out_size, void* d_ws, size_t ws_size,
                              hipStream_t stream)
{
  (void)in_sizes; (void)n_in; (void)out_size; (void)ws_size;
  const float* h   = (const float*)d_in[0];
  const float* pos = (const float*)d_in[1];
  const float* Wq  = (const float*)d_in[2];
  const float* bq  = (const float*)d_in[3];
  const float* Wk  = (const float*)d_in[4];
  const float* bk  = (const float*)d_in[5];
  const float* Wv  = (const float*)d_in[6];
  const float* bv  = (const float*)d_in[7];
  float* out = (float*)d_out;

  // Workspace: Q, K, V intermediates (3 * 8*4096*64 * 4B = 24 MB).
  const size_t per = (size_t)BB * NN * DD;
  float* Qm = (float*)d_ws;
  float* Km = Qm + per;
  float* Vm = Km + per;

  dim3 g1(NN / 64, BB);
  hipLaunchKernelGGL(qkv_proj_wmma, g1, dim3(128), 0, stream,
                     h, Wq, bq, Wk, bk, Wv, bv, Qm, Km, Vm);

  dim3 g2(NN / 256, BB);
  hipLaunchKernelGGL(knn_attn, g2, dim3(256), 0, stream,
                     pos, Qm, Km, Vm, out);
}
